// Patcher_88880053223665
// MI455X (gfx1250) — matrix operand order repaired and checked by reading, not compiled, not run on hardware
//
#include <hip/hip_runtime.h>
#include <cstdint>
#include <cstddef>

// Problem dims (match reference)
#define B_  4
#define T_  2048
#define S_  512
#define D_  768
#define H_  4
#define HD_ 192
#define NT_ (B_*T_)   // 8192 rows
#define NS_ (B_*S_)   // 2048 rows

typedef unsigned short us;   // bf16 storage

// ---------------- types for WMMA ----------------
typedef __bf16          v16bf  __attribute__((ext_vector_type(16)));
typedef float           v8f    __attribute__((ext_vector_type(8)));
typedef unsigned short  u16v8  __attribute__((ext_vector_type(8)));
typedef unsigned short  u16v16 __attribute__((ext_vector_type(16)));

static __device__ __forceinline__ us f2bf(float f) {
  unsigned int u = __float_as_uint(f);
  u += 0x7FFFu + ((u >> 16) & 1u);      // RNE
  return (us)(u >> 16);
}
static __device__ __forceinline__ float bf2f(us u) {
  return __uint_as_float(((unsigned)u) << 16);
}
static __device__ __forceinline__ unsigned pk2bf(float a, float b) {
#if __has_builtin(__builtin_amdgcn_cvt_pk_bf16_f32)
  auto p = __builtin_amdgcn_cvt_pk_bf16_f32(a, b);
  return __builtin_bit_cast(unsigned, p);
#else
  return (unsigned)f2bf(a) | ((unsigned)f2bf(b) << 16);
#endif
}
static __device__ __forceinline__ v16bf mkfrag(u16v8 lo, u16v8 hi) {
  u16v16 w = __builtin_shufflevector(lo, hi, 0,1,2,3,4,5,6,7,8,9,10,11,12,13,14,15);
  return __builtin_bit_cast(v16bf, w);
}
static __device__ __forceinline__ float gelu_tanh(float x) {
  float x3 = x * x * x;
  return 0.5f * x * (1.0f + tanhf(0.7978845608028654f * (x + 0.044715f * x3)));
}

// ---------------- async global->LDS copy (CDNA5 ASYNCcnt path) ----------------
#if __has_builtin(__builtin_amdgcn_global_load_async_to_lds_b128)
#define ASYNC_G2L 1
typedef int vi4 __attribute__((vector_size(16)));
typedef __attribute__((address_space(3))) vi4* lds_vi4p;
static __device__ __forceinline__ void g2l16(void* l, const void* g) {
  __builtin_amdgcn_global_load_async_to_lds_b128((vi4*)g, (lds_vi4p)l, 0, 0);
}
template <int N>
static __device__ __forceinline__ void async_wait_n() {
  asm volatile("s_wait_asynccnt %0" :: "n"(N) : "memory");
}
#else
#define ASYNC_G2L 0
static __device__ __forceinline__ void g2l16(void* l, const void* g) {
  *(uint4*)l = *(const uint4*)g;
}
template <int N>
static __device__ __forceinline__ void async_wait_n() {}
#endif

// ======================================================================
// GEMM: C[Nr,Mc] = act(A[Nr,K] @ W[K,Mc]) (+Res)
// A: bf16 [Nr,K]; Wt: bf16 pre-transposed [Mc][K].
// Tile: 128 rows x (NB*32) cols x 32 K; double-buffered async LDS staging:
// next tile's copies are kept in flight (s_wait_asynccnt <= NPT) while the
// current tile is consumed by WMMA. 8 waves, each computing 32 x (NB*16).
// Output: f32 (Cf, optional Res) or bf16 (Cb).
// ======================================================================
template <int NB>
__global__ __launch_bounds__(256) void gemm_wmma(
    const us* __restrict__ A, int lda,
    const us* __restrict__ Wt,
    float* Cf, const float* Res, us* Cb, int ldc,
    int K, int act)
{
  __shared__ __attribute__((aligned(16))) us As[2][128][40];
  __shared__ __attribute__((aligned(16))) us Bs[2][NB * 32][40];
  const int tid = threadIdx.x;
  const int rowBase = blockIdx.y * 128;
  const int colBase = blockIdx.x * (NB * 32);
  const int w  = tid >> 5, lane = tid & 31;
  const int wr = w & 3,   wc   = w >> 2;
  const int lm = lane & 15, hi = lane >> 4;
  // per-wave async instructions per staged tile
  constexpr int NPT = 2 + (NB * 32 * 4) / 256;

  v8f acc[2][NB];
#pragma unroll
  for (int fm = 0; fm < 2; ++fm)
#pragma unroll
    for (int fn = 0; fn < NB; ++fn) acc[fm][fn] = (v8f)0.f;

  auto stage = [&](int kb, int p) {
#pragma unroll
    for (int c = tid; c < 128 * 4; c += 256) {        // A tile: 128 x 32 bf16
      int r = c >> 2, ch = c & 3;
      g2l16(&As[p][r][ch * 8], A + (size_t)(rowBase + r) * lda + kb + ch * 8);
    }
#pragma unroll
    for (int c = tid; c < NB * 32 * 4; c += 256) {    // B tile: BN x 32 bf16
      int r = c >> 2, ch = c & 3;
      g2l16(&Bs[p][r][ch * 8], Wt + (size_t)(colBase + r) * K + kb + ch * 8);
    }
  };

  const int nsteps = K / 32;
  stage(0, 0);
  for (int i = 0; i < nsteps; ++i) {
    const int p = i & 1;
    __syncthreads();                       // all waves done reading buffer p^1
    if (i + 1 < nsteps) {
      stage((i + 1) * 32, p ^ 1);          // overlap next tile with compute
      async_wait_n<NPT>();                 // current tile (oldest NPT) resident
    } else {
      async_wait_n<0>();
    }
    __syncthreads();

    v16bf a[2], b[NB];
#pragma unroll
    for (int fm = 0; fm < 2; ++fm) {        // A frag: K {koff..+7, koff+16..+23}
      int m = wr * 32 + fm * 16 + lm;
      u16v8 lo = *(const u16v8*)&As[p][m][hi * 8];
      u16v8 h8 = *(const u16v8*)&As[p][m][hi * 8 + 16];
      a[fm] = mkfrag(lo, h8);
    }
#pragma unroll
    for (int fn = 0; fn < NB; ++fn) {       // B frag: K {koff..+7, koff+16..+23}
      int c = wc * (NB * 16) + fn * 16 + lm;
      u16v8 lo = *(const u16v8*)&Bs[p][c][hi * 8];
      u16v8 h8 = *(const u16v8*)&Bs[p][c][hi * 8 + 16];
      b[fn] = mkfrag(lo, h8);
    }
#pragma unroll
    for (int fm = 0; fm < 2; ++fm)
#pragma unroll
      for (int fn = 0; fn < NB; ++fn)
        acc[fm][fn] = __builtin_amdgcn_wmma_f32_16x16x32_bf16(
            false, a[fm], false, b[fn], (short)0, acc[fm][fn], false, false);
  }

  // ---- epilogue ----
#pragma unroll
  for (int fm = 0; fm < 2; ++fm)
#pragma unroll
    for (int fn = 0; fn < NB; ++fn)
#pragma unroll
      for (int r = 0; r < 8; ++r) {
        int grow = rowBase + wr * 32 + fm * 16 + r + hi * 8;
        int gcol = colBase + wc * (NB * 16) + fn * 16 + lm;
        float v = acc[fm][fn][r];
        if (act) v = gelu_tanh(v);
        if (Cb) {
          Cb[(size_t)grow * ldc + gcol] = f2bf(v);
        } else {
          if (Res) v += Res[(size_t)grow * ldc + gcol];
          Cf[(size_t)grow * ldc + gcol] = v;
        }
      }
}

// ======================================================================
// Flash attention (causal), hd=192, bf16 in / bf16 out, fully WMMA.
// One workgroup = 8 waves = 128 query rows of one (b,h).
// K tile async-copied to LDS (row-major); V tile staged transposed.
// ======================================================================
__global__ __launch_bounds__(256) void attn_flash(
    const us* __restrict__ Q, int ldq,
    const us* __restrict__ Kp, int ldk,
    const us* __restrict__ Vp, int ldv,
    us* __restrict__ O, int ldo, int Tb)
{
  __shared__ __attribute__((aligned(16))) us kt[32][200];   // [key][d]
  __shared__ __attribute__((aligned(16))) us vt[192][40];   // [d][key]
  __shared__ __attribute__((aligned(16))) us pt[8][16][40]; // per-wave P
  const int tid = threadIdx.x;
  const int w = tid >> 5, lane = tid & 31, lm = lane & 15, hi = lane >> 4;
  const int b = blockIdx.y / H_, h = blockIdx.y % H_;
  const int qblock = blockIdx.x * 128;
  const int qrow = qblock + w * 16 + lm;
  const float scale = 0.07216878364870322f;   // 1/sqrt(192)

  // --- Q fragments: direct 16B loads (bf16 in memory) ---
  v16bf qf[6];
  const us* qptr = Q + (size_t)(b * Tb + qrow) * ldq + h * HD_;
#pragma unroll
  for (int c = 0; c < 6; ++c) {
    u16v8 lo = *(const u16v8*)(qptr + c * 32 + hi * 8);
    u16v8 h8 = *(const u16v8*)(qptr + c * 32 + hi * 8 + 16);
    qf[c] = mkfrag(lo, h8);
  }

  v8f o[12];
#pragma unroll
  for (int n = 0; n < 12; ++n) o[n] = (v8f)0.f;
  float m_run[8], l_run[8];
#pragma unroll
  for (int r = 0; r < 8; ++r) { m_run[r] = -3.0e38f; l_run[r] = 0.f; }

  const int kend = qblock + 128;
  for (int kb = 0; kb < kend; kb += 32) {
    __syncthreads();
    // K tile: 32 rows x 384B, async 16B chunks
    for (int c = tid; c < 32 * 24; c += 256) {
      int key = c / 24, ch = c % 24;
      g2l16(&kt[key][ch * 8],
            Kp + (size_t)(b * Tb + kb + key) * ldk + h * HD_ + ch * 8);
    }
    // V tile: transpose-staged (bf16 copy, no conversion)
    for (int e = tid; e < 32 * 192; e += 256) {
      int key = e / 192, d = e % 192;
      vt[d][key] = Vp[(size_t)(b * Tb + kb + key) * ldv + h * HD_ + d];
    }
    async_wait_n<0>();
    __syncthreads();

    // ---- S = Q K^T (2x 16-key subtiles) ----
    v8f s[2];
    s[0] = (v8f)0.f; s[1] = (v8f)0.f;
#pragma unroll
    for (int sub = 0; sub < 2; ++sub) {
      int key = sub * 16 + lm;
#pragma unroll
      for (int c = 0; c < 6; ++c) {
        u16v8 lo = *(const u16v8*)&kt[key][c * 32 + hi * 8];
        u16v8 h8 = *(const u16v8*)&kt[key][c * 32 + hi * 8 + 16];
        v16bf bk = mkfrag(lo, h8);
        s[sub] = __builtin_amdgcn_wmma_f32_16x16x32_bf16(
            false, qf[c], false, bk, (short)0, s[sub], false, false);
      }
    }
    // ---- scale + causal mask ----
#pragma unroll
    for (int sub = 0; sub < 2; ++sub) {
      int key = kb + sub * 16 + lm;
#pragma unroll
      for (int r = 0; r < 8; ++r) {
        int mrow = qblock + w * 16 + r + hi * 8;
        s[sub][r] = (key > mrow) ? -1e9f : s[sub][r] * scale;
      }
    }
    // ---- online softmax (16-lane row reductions) ----
#pragma unroll
    for (int r = 0; r < 8; ++r) {
      float v = fmaxf(s[0][r], s[1][r]);
#pragma unroll
      for (int off = 1; off < 16; off <<= 1) v = fmaxf(v, __shfl_xor(v, off, 32));
      float mn = fmaxf(m_run[r], v);
      float f  = __expf(m_run[r] - mn);
      float p0 = __expf(s[0][r] - mn);
      float p1 = __expf(s[1][r] - mn);
      s[0][r] = p0; s[1][r] = p1;
      float t = p0 + p1;
#pragma unroll
      for (int off = 1; off < 16; off <<= 1) t += __shfl_xor(t, off, 32);
      l_run[r] = l_run[r] * f + t;
      m_run[r] = mn;
#pragma unroll
      for (int n = 0; n < 12; ++n) o[n][r] *= f;
    }
    // ---- P: C-frag -> A-frag via per-wave LDS ----
#pragma unroll
    for (int sub = 0; sub < 2; ++sub)
#pragma unroll
      for (int r = 0; r < 8; ++r)
        pt[w][r + hi * 8][sub * 16 + lm] = f2bf(s[sub][r]);
    u16v8 plo = *(const u16v8*)&pt[w][lm][hi * 8];
    u16v8 phi = *(const u16v8*)&pt[w][lm][hi * 8 + 16];
    v16bf pa = mkfrag(plo, phi);
    // ---- O += P V ----
#pragma unroll
    for (int n = 0; n < 12; ++n) {
      int d = n * 16 + lm;
      u16v8 vlo = *(const u16v8*)&vt[d][hi * 8];
      u16v8 vhi = *(const u16v8*)&vt[d][hi * 8 + 16];
      v16bf vb = mkfrag(vlo, vhi);
      o[n] = __builtin_amdgcn_wmma_f32_16x16x32_bf16(
          false, pa, false, vb, (short)0, o[n], false, false);
    }
  }
  // ---- normalize + write bf16 ----
#pragma unroll
  for (int n = 0; n < 12; ++n)
#pragma unroll
    for (int r = 0; r < 8; ++r) {
      int mrow = qblock + w * 16 + r + hi * 8;
      O[(size_t)(b * Tb + mrow) * ldo + h * HD_ + n * 16 + lm] =
          f2bf(o[n][r] / l_run[r]);
    }
}

// ======================================================================
// Weight-only LayerNorm; f32 in, f32 or bf16 out, optional fused +pos.
// ======================================================================
__global__ __launch_bounds__(256) void ln_kernel(
    const float* __restrict__ X, int ldx,
    const float* __restrict__ Wt,
    float* Yf, us* Yb, int ldy,
    const float* __restrict__ Pos, int modT,
    int len, float eps)
{
  __shared__ float red0[256];
  __shared__ float red1[256];
  int row = blockIdx.x;
  const float* xr = X + (size_t)row * ldx;
  const float* pr = Pos ? Pos + (size_t)(row % modT) * len : nullptr;
  float s = 0.f, sq = 0.f;
  for (int i = threadIdx.x; i < len; i += 256) {
    float v = xr[i]; if (pr) v += pr[i];
    s += v; sq += v * v;
  }
  red0[threadIdx.x] = s; red1[threadIdx.x] = sq;
  __syncthreads();
  for (int st = 128; st > 0; st >>= 1) {
    if (threadIdx.x < st) {
      red0[threadIdx.x] += red0[threadIdx.x + st];
      red1[threadIdx.x] += red1[threadIdx.x + st];
    }
    __syncthreads();
  }
  float mean = red0[0] / len;
  float var  = red1[0] / len - mean * mean;
  float rstd = rsqrtf(var + eps);
  for (int i = threadIdx.x * 2; i < len; i += 512) {
    float v0 = xr[i], v1 = xr[i + 1];
    if (pr) { v0 += pr[i]; v1 += pr[i + 1]; }
    v0 = (v0 - mean) * rstd * Wt[i];
    v1 = (v1 - mean) * rstd * Wt[i + 1];
    if (Yb) *(unsigned*)&Yb[(size_t)row * ldy + i] = pk2bf(v0, v1);
    else { Yf[(size_t)row * ldy + i] = v0; Yf[(size_t)row * ldy + i + 1] = v1; }
  }
}

// Weight pre-transpose + convert: Wt[m*K+k] = bf16(Wf[k*Mc+m])
__global__ void wconv_t(const float* __restrict__ Wf, us* __restrict__ Wt,
                        int K, int Mc)
{
  size_t i = (size_t)blockIdx.x * 256 + threadIdx.x;
  size_t n = (size_t)(K / 2) * Mc;
  if (i < n) {
    int m  = (int)(i / (K / 2));
    int k2 = (int)(i % (K / 2)) * 2;
    unsigned v = pk2bf(Wf[(size_t)k2 * Mc + m], Wf[(size_t)(k2 + 1) * Mc + m]);
    *(unsigned*)&Wt[(size_t)m * K + k2] = v;
  }
}

// f32 -> bf16 elementwise (paired)
__global__ void cast_bf(const float* __restrict__ in, us* __restrict__ out,
                        size_t n2)
{
  size_t i = (size_t)blockIdx.x * 256 + threadIdx.x;
  if (i < n2) *(unsigned*)&out[i * 2] = pk2bf(in[i * 2], in[i * 2 + 1]);
}

// gate[n] = dot(Hh[n,:192] (bf16), w2) + b2 ; one wave per row
__global__ __launch_bounds__(256) void gate_dot(
    const us* __restrict__ Hh, const float* __restrict__ w2,
    const float* __restrict__ b2, float* __restrict__ gate)
{
  int w = threadIdx.x >> 5, lane = threadIdx.x & 31;
  int row = blockIdx.x * 8 + w;
  float v = 0.f;
#pragma unroll
  for (int i = 0; i < 6; ++i) {
    int c = lane + 32 * i;
    v += bf2f(Hh[(size_t)row * 192 + c]) * w2[c];
  }
#pragma unroll
  for (int off = 1; off < 32; off <<= 1) v += __shfl_xor(v, off, 32);
  if (lane == 0) gate[row] = v + b2[0];
}

// Exact top-S, stable tie-break (lower index wins), indices emitted ascending.
__global__ __launch_bounds__(256) void topk_kernel(
    const float* __restrict__ gate, int* __restrict__ idx)
{
  __shared__ float gv[T_];
  __shared__ int   fl[T_];
  int b = blockIdx.x;
  for (int i = threadIdx.x; i < T_; i += 256) gv[i] = gate[b * T_ + i];
  __syncthreads();
  for (int i = threadIdx.x; i < T_; i += 256) {
    float x = gv[i]; int rank = 0;
    for (int j = 0; j < T_; ++j) {
      float y = gv[j];
      rank += (y > x) || (y == x && j < i);
    }
    fl[i] = (rank < S_) ? 1 : 0;
  }
  __syncthreads();
  for (int i = threadIdx.x; i < T_; i += 256) {
    if (fl[i]) {
      int pos = 0;
      for (int j = 0; j < i; ++j) pos += fl[j];
      idx[b * S_ + pos] = i;
    }
  }
}

// sel[b,s,:] = bf16( rms(g[b,idx,:]) * sigmoid(gate_ln[b,idx]) )
__global__ __launch_bounds__(256) void gather_rms(
    const float* __restrict__ G, const float* __restrict__ gate,
    const int* __restrict__ idx, us* __restrict__ sel)
{
  __shared__ float red[256];
  __shared__ float wsh;
  int r = blockIdx.x, b = r / S_;
  int t = idx[r];
  const float* gr = G + (size_t)(b * T_ + t) * D_;
  float sq = 0.f;
  for (int i = threadIdx.x; i < D_; i += 256) { float v = gr[i]; sq += v * v; }
  red[threadIdx.x] = sq; __syncthreads();
  for (int st = 128; st > 0; st >>= 1) {
    if (threadIdx.x < st) red[threadIdx.x] += red[threadIdx.x + st];
    __syncthreads();
  }
  if (threadIdx.x == 0) {
    float rms = rsqrtf(red[0] / D_ + 1e-8f);
    float gvv = gate[b * T_ + t];
    wsh = rms / (1.f + __expf(-gvv));
  }
  __syncthreads();
  float wv = wsh;
  for (int i = threadIdx.x * 2; i < D_; i += 512)
    *(unsigned*)&sel[(size_t)r * D_ + i] = pk2bf(gr[i] * wv, gr[i + 1] * wv);
}

__global__ void fill_pos(const float* __restrict__ pos, float* __restrict__ out)
{
  size_t i = (size_t)blockIdx.x * blockDim.x + threadIdx.x;
  if (i < (size_t)NT_ * D_) {
    int d = (int)(i % D_);
    int t = (int)((i / D_) % T_);
    out[i] = pos[(size_t)t * D_ + d];
  }
}

__global__ __launch_bounds__(256) void scatter_add(
    const float* __restrict__ xd, const int* __restrict__ idx,
    float* __restrict__ scat)
{
  int r = blockIdx.x, b = r / S_;
  int t = idx[r];
  for (int i = threadIdx.x; i < D_; i += 256)
    scat[(size_t)(b * T_ + t) * D_ + i] += xd[(size_t)r * D_ + i];
}

// ======================================================================
// Host orchestration
// ======================================================================
struct BlockAll {
  const float* ln1; const us* attnwT; const us* projwT;
  const float* ln2; const us* fcwT;   const us* fc2wT;
};

static inline void launch_gemm(const us* A, int lda, const us* Wt,
                               float* Cf, const float* Res, us* Cb, int ldc,
                               int Nr, int K, int Mc, int act, hipStream_t st) {
  if (Mc % 128 == 0) {
    dim3 g(Mc / 128, Nr / 128);
    gemm_wmma<4><<<g, 256, 0, st>>>(A, lda, Wt, Cf, Res, Cb, ldc, K, act);
  } else {
    dim3 g(Mc / 64, Nr / 128);
    gemm_wmma<2><<<g, 256, 0, st>>>(A, lda, Wt, Cf, Res, Cb, ldc, K, act);
  }
}
static inline void launch_ln(const float* X, int ldx, const float* w,
                             float* Yf, us* Yb, int ldy,
                             const float* pos, int modT,
                             int len, int rows, hipStream_t st) {
  ln_kernel<<<rows, 256, 0, st>>>(X, ldx, w, Yf, Yb, ldy, pos, modT, len, 1e-5f);
}
static void run_block(float* cur, int Tb, const BlockAll& p,
                      us* t0b, us* t2b, us* bigb, hipStream_t st) {
  int Nr = B_ * Tb;
  launch_ln(cur, D_, p.ln1, nullptr, t0b, D_, nullptr, 1, D_, Nr, st);
  launch_gemm(t0b, D_, p.attnwT, nullptr, nullptr, bigb, 3 * D_, Nr, D_, 3 * D_, 0, st);
  dim3 ga(Tb / 128, B_ * H_);
  attn_flash<<<ga, 256, 0, st>>>(bigb, 3 * D_, bigb + D_, 3 * D_,
                                 bigb + 2 * D_, 3 * D_, t2b, D_, Tb);
  launch_gemm(t2b, D_, p.projwT, cur, cur, nullptr, D_, Nr, D_, D_, 0, st);
  launch_ln(cur, D_, p.ln2, nullptr, t0b, D_, nullptr, 1, D_, Nr, st);
  launch_gemm(t0b, D_, p.fcwT, nullptr, nullptr, bigb, 4 * D_, Nr, D_, 4 * D_, 1, st);
  launch_gemm(bigb, 4 * D_, p.fc2wT, cur, cur, nullptr, D_, Nr, 4 * D_, D_, 0, st);
}

extern "C" void kernel_launch(void* const* d_in, const int* in_sizes, int n_in,
                              void* d_out, int out_size, void* d_ws, size_t ws_size,
                              hipStream_t stream)
{
  (void)in_sizes; (void)n_in; (void)out_size; (void)ws_size;
  const float* x = (const float*)d_in[0];
  auto F = [&](int i) { return (const float*)d_in[i]; };

  // ---- workspace carving (256B-aligned byte cursor) ----
  unsigned char* base = (unsigned char*)d_ws;
  size_t cur_off = 0;
  auto allocB = [&](size_t bytes) {
    unsigned char* p = base + cur_off;
    cur_off += (bytes + 255) & ~(size_t)255;
    return p;
  };
  const size_t BTD = (size_t)NT_ * D_;
  float* acc     = (float*)allocB(BTD * 4);            // residual / g / pds
  float* scat    = (float*)allocB(BTD * 4);            // scattered stream
  float* query   = (float*)allocB(BTD * 4);            // query stream
  float* up      = (float*)allocB((size_t)NS_ * D_ * 4);
  float* xd      = (float*)allocB((size_t)NS_ * D_ * 4);
  float* gate    = (float*)allocB(NT_ * 4);
  float* gate_ln = (float*)allocB(NT_ * 4);
  int*   idx     = (int*)  allocB(NT_ * 4);
  us* t0b   = (us*)allocB(BTD * 2);
  us* t2b   = (us*)allocB(BTD * 2);
  us* bigb  = (us*)allocB((size_t)NT_ * 4 * D_ * 2);
  us* selb  = (us*)allocB((size_t)NS_ * D_ * 2);
  us* gbf   = (us*)allocB(BTD * 2);
  us* scatb = (us*)allocB(BTD * 2);
  us* gateh = (us*)allocB((size_t)NT_ * (D_ / 4) * 2);
  us* wcur  = (us*)allocB(31408128ull * 2);            // all transposed weights
  us* wp = wcur;
  auto conv_w = [&](const float* Wf, int K, int Mc) {
    us* out = wp; wp += (size_t)K * Mc;
    size_t n = (size_t)(K / 2) * Mc;
    wconv_t<<<(unsigned)((n + 255) / 256), 256, 0, stream>>>(Wf, out, K, Mc);
    return (const us*)out;
  };

  // ---- weight pre-transpose/convert (bf16 [M][K]) ----
  BlockAll qb{F(1),  conv_w(F(2), D_, 3*D_), conv_w(F(3), D_, D_),
              F(4),  conv_w(F(5), D_, 4*D_), conv_w(F(6), 4*D_, D_)};
  const us* upg1T = conv_w(F(7), D_, D_/4);
  const float* up_gate_w2 = F(8); const float* up_gate_b2 = F(9);
  const float* gate_norm_w = F(10);
  const us* upprojT = conv_w(F(11), D_, D_);
  BlockAll pb{F(12), conv_w(F(13), D_, 3*D_), conv_w(F(14), D_, D_),
              F(15), conv_w(F(16), D_, 4*D_), conv_w(F(17), 4*D_, D_)};
  const float* down_norm_w = F(18); const float* pos_emb = F(19);
  BlockAll db{F(20), conv_w(F(21), D_, 3*D_), conv_w(F(22), D_, D_),
              F(23), conv_w(F(24), D_, 4*D_), conv_w(F(25), 4*D_, D_)};
  const us* downprojT = conv_w(F(26), D_, D_);
  const float* res_norm_w = F(27);
  const float* lnq_w = F(28); const float* lnkv_w = F(29);
  const us* q_wT  = conv_w(F(30), D_, D_);
  const us* kv_wT = conv_w(F(31), D_, 2*D_);
  const us* c_projT = conv_w(F(32), D_, D_);
  const float* c_ln2 = F(33);
  const us* c_fcT  = conv_w(F(34), D_, 4*D_);
  const us* c_fc2T = conv_w(F(35), 4*D_, D_);
  BlockAll sb{F(36), conv_w(F(37), D_, 3*D_), conv_w(F(38), D_, D_),
              F(39), conv_w(F(40), D_, 4*D_), conv_w(F(41), 4*D_, D_)};

  // ---- abstract_up ----
  (void)hipMemcpyAsync(acc, x, BTD * 4, hipMemcpyDeviceToDevice, stream);
  run_block(acc, T_, qb, t0b, t2b, bigb, stream);                   // acc = g
  cast_bf<<<(unsigned)((BTD / 2 + 255) / 256), 256, 0, stream>>>(acc, gbf, BTD / 2);
  launch_gemm(gbf, D_, upg1T, nullptr, nullptr, gateh, D_/4, NT_, D_, D_/4, 1, stream);
  gate_dot<<<NT_ / 8, 256, 0, stream>>>(gateh, up_gate_w2, up_gate_b2, gate);
  ln_kernel<<<B_, 256, 0, stream>>>(gate, T_, gate_norm_w, gate_ln, nullptr, T_,
                                    nullptr, 1, T_, 1e-5f);
  topk_kernel<<<B_, 256, 0, stream>>>(gate_ln, idx);
  gather_rms<<<NS_, 256, 0, stream>>>(acc, gate_ln, idx, selb);
  launch_gemm(selb, D_, upprojT, up, nullptr, nullptr, D_, NS_, D_, D_, 0, stream);
  run_block(up, S_, pb, t0b, t2b, bigb, stream);                    // proj_block

  // ---- abstract_down ----
  launch_ln(up, D_, down_norm_w, xd, nullptr, D_, nullptr, 1, D_, NS_, stream);
  fill_pos<<<(unsigned)((BTD + 255) / 256), 256, 0, stream>>>(pos_emb, scat);
  scatter_add<<<NS_, 256, 0, stream>>>(xd, idx, scat);
  run_block(scat, T_, db, t0b, t2b, bigb, stream);                  // down_block
  cast_bf<<<(unsigned)((BTD / 2 + 255) / 256), 256, 0, stream>>>(scat, scatb, BTD / 2);
  launch_gemm(scatb, D_, downprojT, acc, nullptr, nullptr, D_, NT_, D_, D_, 0, stream); // acc = pds
  launch_ln(x, D_, res_norm_w, query, nullptr, D_, pos_emb, T_, D_, NT_, stream);

  // ---- CombineBlock (down_scatter2): causal cross-attn ----
  launch_ln(query, D_, lnq_w, nullptr, t0b, D_, nullptr, 1, D_, NT_, stream);
  launch_gemm(t0b, D_, q_wT, nullptr, nullptr, t2b, D_, NT_, D_, D_, 0, stream);       // qh
  launch_ln(acc, D_, lnkv_w, nullptr, t0b, D_, nullptr, 1, D_, NT_, stream);
  launch_gemm(t0b, D_, kv_wT, nullptr, nullptr, bigb, 2 * D_, NT_, D_, 2 * D_, 0, stream);
  {
    dim3 ga(T_ / 128, B_ * H_);
    attn_flash<<<ga, 256, 0, stream>>>(t2b, D_, bigb, 2 * D_, bigb + D_, 2 * D_,
                                       t0b, D_, T_);
  }
  launch_gemm(t0b, D_, c_projT, query, query, nullptr, D_, NT_, D_, D_, 0, stream);
  launch_ln(query, D_, c_ln2, nullptr, t0b, D_, nullptr, 1, D_, NT_, stream);
  launch_gemm(t0b, D_, c_fcT, nullptr, nullptr, bigb, 4 * D_, NT_, D_, 4 * D_, 1, stream);
  launch_gemm(bigb, 4 * D_, c_fc2T, query, query, nullptr, D_, NT_, 4 * D_, D_, 0, stream);

  // ---- down_scan + output ----
  run_block(query, T_, sb, t0b, t2b, bigb, stream);
  (void)hipMemcpyAsync(d_out, query, BTD * 4, hipMemcpyDeviceToDevice, stream);
}
